// VoxelBackBone8x_29274497089669
// MI455X (gfx1250) — matrix operand order repaired and checked by reading, not compiled, not run on hardware
//
#include <hip/hip_runtime.h>
#include <hip/hip_bf16.h>

// ---------------------------------------------------------------------------
// Types for CDNA5 WMMA
// ---------------------------------------------------------------------------
typedef __attribute__((ext_vector_type(16))) __bf16 v16bf;
typedef __attribute__((ext_vector_type(8)))  float  v8f;

static __device__ __forceinline__ unsigned short f32_to_bf16_bits(float f) {
    unsigned int u = __builtin_bit_cast(unsigned int, f);
    unsigned int r = (u + 0x7FFFu + ((u >> 16) & 1u)) >> 16;   // RNE
    return (unsigned short)r;
}

// ---------------------------------------------------------------------------
// Utility kernels
// ---------------------------------------------------------------------------
__global__ void zero_u8_kernel(unsigned char* p, long n) {
    long i = (long)blockIdx.x * blockDim.x + threadIdx.x;
    if (i < n) p[i] = 0;
}

// Build transposed, channel-padded bf16 weight panel:
//   o[n*KPAD + k] = w[(tap*CIN + ci)*cout + n] for tap=k/CPAD, ci=k%CPAD
//   zero when n>=cout, tap>=KVOL, or ci>=CIN (K-space padded to CPAD/tap)
__global__ void wconv_t_kernel(const float* w, unsigned short* o,
                               int KPAD, int CPAD, int CIN, int KVOL,
                               int cout, int NT16) {
    int i = blockIdx.x * blockDim.x + threadIdx.x;
    int total = NT16 * KPAD;
    if (i >= total) return;
    int n = i / KPAD;
    int k = i - n * KPAD;
    int tap = k / CPAD;
    int ci  = k - tap * CPAD;
    unsigned short v = 0;
    if (n < cout && tap < KVOL && ci < CIN)
        v = f32_to_bf16_bits(w[((long)tap * CIN + ci) * cout + n]);
    o[i] = v;
}

__global__ void sb_kernel(const float* gamma, const float* beta,
                          const float* mean, const float* var,
                          float* sc, float* bs, int cout) {
    int i = blockIdx.x * blockDim.x + threadIdx.x;
    if (i < cout) {
        float s = gamma[i] * rsqrtf(var[i] + 1e-3f);
        sc[i] = s;
        bs[i] = beta[i] - mean[i] * s;
    }
}

__global__ void scatter_kernel(const float* feat, const int* coords,
                               unsigned short* x, unsigned char* mask,
                               int n, int D, int H, int W, int C) {
    int i = blockIdx.x * blockDim.x + threadIdx.x;
    if (i >= n) return;
    int b  = coords[4 * i + 0];
    int z  = coords[4 * i + 1];
    int y  = coords[4 * i + 2];
    int xx = coords[4 * i + 3];
    long vox = ((((long)b * D + z) * H + y) * W + xx);
    for (int c = 0; c < C; ++c)
        x[vox * C + c] = f32_to_bf16_bits(feat[(long)i * C + c]);
    mask[vox] = 1;
}

__global__ void mask_update_kernel(const unsigned char* mi, unsigned char* mo,
                                   int inD, int inH, int inW,
                                   int outD, int outH, int outW,
                                   int kd, int kh, int kw,
                                   int sd, int sh, int sw,
                                   int pd, int ph, int pw, int M) {
    int m = blockIdx.x * blockDim.x + threadIdx.x;
    if (m >= M) return;
    int t = m;
    int ow = t % outW; t /= outW;
    int oh = t % outH; t /= outH;
    int od = t % outD; int b = t / outD;
    unsigned char v = 0;
    for (int kz = 0; kz < kd; ++kz) {
        int iz = od * sd - pd + kz;
        if ((unsigned)iz >= (unsigned)inD) continue;
        for (int ky = 0; ky < kh; ++ky) {
            int iy = oh * sh - ph + ky;
            if ((unsigned)iy >= (unsigned)inH) continue;
            for (int kx = 0; kx < kw; ++kx) {
                int ix = ow * sw - pw + kx;
                if ((unsigned)ix >= (unsigned)inW) continue;
                v |= mi[(((long)b * inD + iz) * inH + iy) * (long)inW + ix];
            }
        }
    }
    mo[m] = v ? 1 : 0;
}

// ---------------------------------------------------------------------------
// Implicit-GEMM conv3d + BN + ReLU + mask, bf16 inputs, f32 accum via WMMA.
//   GEMM K-space is tap-major with channels padded to CPAD (power of two,
//   zero weights in the pad) so every 8-consecutive-K run stays inside one
//   tap: A fragment = 2 runs = aligned b128 (CIN%8==0) or 2x b64 loads, tap
//   decode is shift/and only. Block: 256 threads = 8 waves, 128 output rows,
//   each wave owns all NT N-tiles. Per-(row,tap) offsets in LDS; transposed
//   weight panel staged to LDS in 512-wide sections via uint4.
// ---------------------------------------------------------------------------
struct ConvP {
    const unsigned short* in;    // bf16 bits, NDHWC
    const unsigned short* w;     // bf16 bits, transposed panel [NT16][KPAD]
    const float* scale;
    const float* bias;
    const unsigned char* mask;   // per OUTPUT voxel
    unsigned short* outB;        // bf16 output (nullptr on final layer)
    float* outF;                 // f32 output (final layer only)
    int inD, inH, inW, outD, outH, outW;
    int sd, sh, sw, pd, ph, pw;
    int M;
};

template<int CIN, int COUT, int KD, int KH, int KW>
__global__ __launch_bounds__(256) void conv_wmma_kernel(ConvP p) {
    constexpr int KVOL   = KD * KH * KW;
    constexpr int CPAD   = (CIN <= 8) ? 8 : (CIN <= 16) ? 16 : (CIN <= 32) ? 32 : 64;
    constexpr int SHIFT  = (CPAD == 8) ? 3 : (CPAD == 16) ? 4 : (CPAD == 32) ? 5 : 6;
    constexpr int KP     = KVOL * CPAD;          // padded K extent
    constexpr int KPAD   = (KP + 31) & ~31;      // weight panel row length
    constexpr int COUT16 = (COUT + 15) & ~15;
    constexpr int NT     = COUT16 / 16;
    constexpr int KSEC   = (KPAD < 512) ? KPAD : 512;   // static LDS < 64KB
    constexpr int QROW   = KSEC / 8;                    // uint4 per LDS row
    constexpr int ROWS   = 128;                         // M rows per block

    __shared__ __align__(16) unsigned short sWT[COUT16 * KSEC];
    __shared__ int sOff[ROWS * KVOL];

    const int tid    = threadIdx.x;
    const int lane   = tid & 31;
    const int wave   = tid >> 5;
    const int mBase0 = blockIdx.x * ROWS;

    if (tid == 0) __builtin_prefetch(p.in, 0, 0);   // global_prefetch_b8

    // ---- precompute per-(row,tap) input element offsets (-1 = OOB) ----
    for (int e = tid; e < ROWS * KVOL; e += 256) {
        int r   = e / KVOL;            // const divisor
        int tap = e - r * KVOL;
        int row = mBase0 + r;
        int off = -1;
        if (row < p.M) {
            int t  = row;
            int ow = t % p.outW; t /= p.outW;
            int oh = t % p.outH; t /= p.outH;
            int od = t % p.outD; int b = t / p.outD;
            int kz = tap / (KH * KW);
            int r2 = tap - kz * (KH * KW);
            int ky = r2 / KW;
            int kx = r2 - ky * KW;
            int iz = od * p.sd - p.pd + kz;
            int iy = oh * p.sh - p.ph + ky;
            int ix = ow * p.sw - p.pw + kx;
            if ((unsigned)iz < (unsigned)p.inD &&
                (unsigned)iy < (unsigned)p.inH &&
                (unsigned)ix < (unsigned)p.inW)
                off = (((b * p.inD + iz) * p.inH + iy) * p.inW + ix) * CIN;
        }
        sOff[e] = off;
    }

    const int half  = (lane >> 4) << 3;           // A layout K-half select
    const int nn    = lane & 15;                  // B/D column
    const int rbase = (wave * 16 + (lane & 15)) * KVOL;

    v8f acc[NT];
#pragma unroll
    for (int nt = 0; nt < NT; ++nt)
#pragma unroll
        for (int j = 0; j < 8; ++j) acc[nt][j] = 0.f;

    const uint4* gW = (const uint4*)p.w;
    uint4*       lW = (uint4*)sWT;

    for (int ks0 = 0; ks0 < KP; ks0 += KSEC) {
        __syncthreads();   // also covers sOff on first iteration
        // ---- stage KSEC-wide weight panel section (global b128 -> ds b128) ----
        for (int idx = tid; idx < COUT16 * QROW; idx += 256) {
            int rowN = idx / QROW;     // const divisor
            int c    = idx - rowN * QROW;
            lW[idx]  = gW[rowN * (KPAD / 8) + (ks0 / 8) + c];
        }
        __syncthreads();

        const int kend = ((KP - ks0) < KSEC) ? (KP - ks0) : KSEC;
        for (int kkl = 0; kkl < kend; kkl += 32) {
            // ---- A fragment: two runs of 8 consecutive padded-K per lane ----
            union { v16bf v; uint2 q[4]; uint4 qq[2]; } A;
#pragma unroll
            for (int run = 0; run < 2; ++run) {
                const int kb   = ks0 + kkl + run * 16 + half;  // multiple of 8
                const bool inK = kb < KP;                      // pad tail of KPAD
                const int tap0 = kb >> SHIFT;
                const int ci0  = kb & (CPAD - 1);
                const int off  = inK ? sOff[rbase + tap0] : -1;
                if (off >= 0) {
                    // run is 16 contiguous bytes; pad channels hit neighbor
                    // voxel data but multiply against zero weights.
                    if constexpr ((CIN % 8) == 0) {
                        A.qq[run] = *(const uint4*)(p.in + off + ci0);   // b128
                    } else {
                        const uint2* src = (const uint2*)(p.in + off + ci0);
                        A.q[run * 2 + 0] = src[0];                      // b64
                        A.q[run * 2 + 1] = src[1];                      // b64
                    }
                } else {
                    uint4 z; z.x = 0u; z.y = 0u; z.z = 0u; z.w = 0u;
                    A.qq[run] = z;
                }
            }
            // ---- NT x (B fragment via 2x ds_load_b128 + WMMA) ----
            const int kbl = kkl + ((lane >> 4) << 3);
#pragma unroll
            for (int nt = 0; nt < NT; ++nt) {
                union { v16bf v; uint4 q[2]; } Bf;
                const unsigned short* src = &sWT[(nt * 16 + nn) * KSEC + kbl];
                Bf.q[0] = *(const uint4*)(src);
                Bf.q[1] = *(const uint4*)(src + 16);
                acc[nt] = __builtin_amdgcn_wmma_f32_16x16x32_bf16(
                    false, A.v, false, Bf.v, (short)0, acc[nt], false, false);
            }
        }
    }

    // ---- epilogue: BN scale/bias, ReLU, mask, store ----
    const int mBase = mBase0 + wave * 16;
    const int mOff  = (lane >> 4) << 3;
#pragma unroll
    for (int nt = 0; nt < NT; ++nt) {
        int n = nt * 16 + nn;
        if ((COUT % 16) != 0 && n >= COUT) continue;
        float sc = p.scale[n], bs = p.bias[n];
#pragma unroll
        for (int r = 0; r < 8; ++r) {
            int m = mBase + mOff + r;
            if (m < p.M) {
                float v = acc[nt][r] * sc + bs;
                v = fmaxf(v, 0.f);
                v *= (float)p.mask[m];
                long o = (long)m * COUT + n;
                if (p.outF) p.outF[o] = v;
                else        p.outB[o] = f32_to_bf16_bits(v);
            }
        }
    }
}

// ---------------------------------------------------------------------------
// Host orchestration
// ---------------------------------------------------------------------------
extern "C" void kernel_launch(void* const* d_in, const int* in_sizes, int n_in,
                              void* d_out, int out_size, void* d_ws, size_t ws_size,
                              hipStream_t stream) {
    (void)in_sizes; (void)n_in; (void)out_size; (void)ws_size;

    constexpr int Bn = 2, NVOX = 20000, CIN0 = 4;
    constexpr int Z0 = 41, Y0 = 200, X0 = 176;

    // layers: cin,cout, kd,kh,kw, sd,sh,sw, pd,ph,pw, subm
    static const int L[12][12] = {
        { 4,12, 3,3,3, 1,1,1, 1,1,1, 1},
        {12,12, 3,3,3, 1,1,1, 1,1,1, 1},
        {12,24, 3,3,3, 2,2,2, 1,1,1, 0},
        {24,24, 3,3,3, 1,1,1, 1,1,1, 1},
        {24,24, 3,3,3, 1,1,1, 1,1,1, 1},
        {24,48, 3,3,3, 2,2,2, 1,1,1, 0},
        {48,48, 3,3,3, 1,1,1, 1,1,1, 1},
        {48,48, 3,3,3, 1,1,1, 1,1,1, 1},
        {48,48, 3,3,3, 2,2,2, 0,1,1, 0},
        {48,48, 3,3,3, 1,1,1, 1,1,1, 1},
        {48,48, 3,3,3, 1,1,1, 1,1,1, 1},
        {48,96, 3,1,1, 2,1,1, 0,0,0, 0},
    };

    const float* feat   = (const float*)d_in[0];
    const int*   coords = (const int*)d_in[1];

    // ---- carve workspace ----
    char* ws = (char*)d_ws;
    size_t off = 0;
    auto carve = [&](size_t bytes) -> char* {
        off = (off + 255) & ~(size_t)255;
        char* p = ws + off;
        off += bytes;
        return p;
    };
    const size_t MAX_ACT = (size_t)Bn * Z0 * Y0 * X0 * 12;  // 34.6M elems
    const size_t MAX_VOX = (size_t)Bn * Z0 * Y0 * X0;       // 2.89M voxels
    unsigned short* actA = (unsigned short*)carve(MAX_ACT * 2);
    unsigned short* actB = (unsigned short*)carve(MAX_ACT * 2);
    unsigned char*  mskA = (unsigned char*)carve(MAX_VOX);
    unsigned char*  mskB = (unsigned char*)carve(MAX_VOX);
    unsigned short* wbuf = (unsigned short*)carve(4u << 20);
    float*          sbuf = (float*)carve(12 * 256 * sizeof(float));

    // ---- per-layer transposed, channel-padded bf16 weight panels + BN ----
    unsigned short* wL[12];
    float *scL[12], *bsL[12];
    {
        size_t woff = 0;
        for (int i = 0; i < 12; ++i) {
            int cin  = L[i][0], cout = L[i][1];
            int kvol = L[i][2] * L[i][3] * L[i][4];
            int CPAD = (cin <= 8) ? 8 : (cin <= 16) ? 16 : (cin <= 32) ? 32 : 64;
            int KP   = kvol * CPAD;
            int KPAD = (KP + 31) & ~31;
            int NT16 = (cout + 15) & ~15;
            wL[i] = wbuf + woff;
            woff += (size_t)NT16 * KPAD;
            const float* w     = (const float*)d_in[2 + 5 * i + 0];
            const float* gamma = (const float*)d_in[2 + 5 * i + 1];
            const float* beta  = (const float*)d_in[2 + 5 * i + 2];
            const float* mean  = (const float*)d_in[2 + 5 * i + 3];
            const float* var   = (const float*)d_in[2 + 5 * i + 4];
            int total = NT16 * KPAD;
            wconv_t_kernel<<<(total + 255) / 256, 256, 0, stream>>>(
                w, wL[i], KPAD, CPAD, cin, kvol, cout, NT16);
            scL[i] = sbuf + i * 256;
            bsL[i] = sbuf + i * 256 + 128;
            sb_kernel<<<1, 128, 0, stream>>>(gamma, beta, mean, var, scL[i], bsL[i], cout);
        }
    }

    // ---- zero dense input + mask, scatter voxels ----
    {
        long nbytes = (long)Bn * Z0 * Y0 * X0 * CIN0 * 2;
        zero_u8_kernel<<<(unsigned)((nbytes + 255) / 256), 256, 0, stream>>>(
            (unsigned char*)actA, nbytes);
        long mbytes = (long)MAX_VOX;
        zero_u8_kernel<<<(unsigned)((mbytes + 255) / 256), 256, 0, stream>>>(mskA, mbytes);
        int n = Bn * NVOX;
        scatter_kernel<<<(n + 255) / 256, 256, 0, stream>>>(
            feat, coords, actA, mskA, n, Z0, Y0, X0, CIN0);
    }

    // ---- run layers ----
    unsigned short* cur = actA;
    unsigned short* nxt = actB;
    unsigned char*  curM = mskA;
    unsigned char*  nxtM = mskB;
    int D = Z0, H = Y0, W = X0;

    for (int i = 0; i < 12; ++i) {
        int kd = L[i][2], kh = L[i][3], kw = L[i][4];
        int sd = L[i][5], sh = L[i][6], sw = L[i][7];
        int pd = L[i][8], ph = L[i][9], pw = L[i][10];
        int subm = L[i][11];

        int oD = subm ? D : (D + 2 * pd - kd) / sd + 1;
        int oH = subm ? H : (H + 2 * ph - kh) / sh + 1;
        int oW = subm ? W : (W + 2 * pw - kw) / sw + 1;
        int M = Bn * oD * oH * oW;

        if (!subm) {
            mask_update_kernel<<<(M + 255) / 256, 256, 0, stream>>>(
                curM, nxtM, D, H, W, oD, oH, oW,
                kd, kh, kw, sd, sh, sw, pd, ph, pw, M);
            unsigned char* t = curM; curM = nxtM; nxtM = t;
        }

        ConvP p;
        p.in = cur; p.w = wL[i]; p.scale = scL[i]; p.bias = bsL[i];
        p.mask = curM;
        p.outB = (i == 11) ? nullptr : nxt;
        p.outF = (i == 11) ? (float*)d_out : nullptr;
        p.inD = D; p.inH = H; p.inW = W;
        p.outD = oD; p.outH = oH; p.outW = oW;
        p.sd = sd; p.sh = sh; p.sw = sw;
        p.pd = pd; p.ph = ph; p.pw = pw;
        p.M = M;

        dim3 grid((unsigned)((M + 127) / 128));
        if      (i == 0)            conv_wmma_kernel< 4,12,3,3,3><<<grid,256,0,stream>>>(p);
        else if (i == 1)            conv_wmma_kernel<12,12,3,3,3><<<grid,256,0,stream>>>(p);
        else if (i == 2)            conv_wmma_kernel<12,24,3,3,3><<<grid,256,0,stream>>>(p);
        else if (i == 3 || i == 4)  conv_wmma_kernel<24,24,3,3,3><<<grid,256,0,stream>>>(p);
        else if (i == 5)            conv_wmma_kernel<24,48,3,3,3><<<grid,256,0,stream>>>(p);
        else if (i <= 10)           conv_wmma_kernel<48,48,3,3,3><<<grid,256,0,stream>>>(p);
        else                        conv_wmma_kernel<48,96,3,1,1><<<grid,256,0,stream>>>(p);

        unsigned short* t = cur; cur = nxt; nxt = t;
        D = oD; H = oH; W = oW;
    }
}
